// MultiheadAttention_12257836663070
// MI455X (gfx1250) — matrix operand order repaired and checked by reading, not compiled, not run on hardware
//
#include <hip/hip_runtime.h>
#include <hip/hip_bf16.h>

// ---------------------------------------------------------------------------
// MultiheadAttention forward for MI455X (gfx1250, wave32, WMMA bf16->f32)
// B=1, S=4096, E=1024, H=16, HD=64. Causal mask synthesized (input mask unused).
// Data movement: TDM (tensor_load_to_lds) for K tiles, vector loads + LDS
// fragment-major swizzle for B/V tiles, contiguous v16bf fragment loads.
// ---------------------------------------------------------------------------

typedef __attribute__((ext_vector_type(16))) __bf16 v16bf;
typedef __attribute__((ext_vector_type(8)))  __bf16 v8bf;
typedef __attribute__((ext_vector_type(8)))  float  v8f;
typedef __attribute__((ext_vector_type(8)))  float  v8fv;
typedef __attribute__((ext_vector_type(4)))  unsigned int u32x4;
typedef __attribute__((ext_vector_type(8)))  int    i32x8;
typedef __attribute__((ext_vector_type(4)))  int    i32x4;

#define S_LEN  4096
#define EMB    1024
#define NH     16
#define HDIM   64
#define DMODEL 1024  // NH * HDIM

__device__ __forceinline__ v8f wmma_bf16f32(v16bf a, v16bf b, v8f c) {
  return __builtin_amdgcn_wmma_f32_16x16x32_bf16(
      /*neg_a=*/false, a, /*neg_b=*/false, b,
      /*c_mod=*/(short)0, c, /*reuse_a=*/false, /*reuse_b=*/false);
}

// A-fragment (16x32 bf16, row-major, leading dim ld). Lane's 16 values are
// one contiguous run: K = 16*half + {0..15}.
__device__ __forceinline__ v16bf load_a_frag(const __bf16* base, int ld) {
  const int lane = threadIdx.x & 31;
  const __bf16* row = base + (lane & 15) * ld + 16 * (lane >> 4);
  const v8bf lo = *(const v8bf*)row;         // K 0..7   (+16*half)
  const v8bf hi = *(const v8bf*)(row + 8);   // K 8..15  (+16*half)
  return __builtin_shufflevector(lo, hi, 0, 1, 2, 3, 4, 5, 6, 7,
                                 8, 9, 10, 11, 12, 13, 14, 15);
}

// B-fragment for Q*K^T from row-major K tile: B[k=hd][n=key] = Kt[key][hd],
// lane needs key=lane&15, hd = 16*half + {0..15}: one contiguous 32B run.
__device__ __forceinline__ v16bf load_bT_frag(const __bf16* base, int row_stride) {
  const int lane = threadIdx.x & 31;
  return *(const v16bf*)(base + (lane & 15) * row_stride + 16 * (lane >> 4));
}

// B-fragment from fragment-major swizzled LDS: sw[t][lane][16] contiguous.
__device__ __forceinline__ v16bf load_b_frag_sw(const __bf16* sw, int t) {
  const int lane = threadIdx.x & 31;
  return *(const v16bf*)(sw + ((t * 32 + lane) << 4));
}

// 8-element row staging: fp32 -> packed bf16 convert, or raw bf16 copy.
__device__ __forceinline__ v8bf cvt8(const float* s) {
  return __builtin_convertvector(*(const v8fv*)s, v8bf);
}
__device__ __forceinline__ v8bf cvt8(const __bf16* s) { return *(const v8bf*)s; }

__device__ __forceinline__ void stage8(const float* s, __bf16* d) {
  *(v8bf*)d = cvt8(s);
}
__device__ __forceinline__ void stage8(const __bf16* s, __bf16* d) {
  *(v8bf*)d = cvt8(s);
}

// Stage a 32 x tileW tile from row-major W (ld ldw) into fragment-major LDS:
// sw[(n>>4)][16*(k>>4) + (n&15)][k&15]. Each thread handles a 2(k) x 8(n)
// patch so (k,k+1) pairs merge into 4-byte LDS stores.
template <typename T>
__device__ __forceinline__ void stage_b_swz(const T* W, int ldw, __bf16* sw,
                                            int tileW, int tid) {
  const int npatch = tileW / 8;
  const int k2 = (tid / npatch) * 2;
  const int n0 = (tid % npatch) * 8;
  const v8bf r0 = cvt8(W + (size_t)k2 * ldw + n0);
  const v8bf r1 = cvt8(W + (size_t)(k2 + 1) * ldw + n0);
#pragma unroll
  for (int i = 0; i < 8; ++i) {
    const int n = n0 + i;
    const int lane = ((k2 >> 4) << 4) | (n & 15);
    __bf16* d = sw + ((((n >> 4) * 32 + lane) << 4) + (k2 & 15));
    d[0] = r0[i];
    d[1] = r1[i];
  }
}

// Row reductions across the 16-lane half holding one D-matrix row (wave32).
__device__ __forceinline__ float rmax16(float v) {
#pragma unroll
  for (int o = 1; o < 16; o <<= 1) v = fmaxf(v, __shfl_xor(v, o, 16));
  return v;
}
__device__ __forceinline__ float rsum16(float v) {
#pragma unroll
  for (int o = 1; o < 16; o <<= 1) v += __shfl_xor(v, o, 16);
  return v;
}

// ---------------------------------------------------------------------------
// Tensor Data Mover: 2-D bf16 tile load, global -> LDS. Descriptor bitfields
// per CDNA5 ISA (group 0) / §8.4 (group 1):
//  g0: [1:0]count=1 | [63:32]lds_addr | [120:64]global_addr | [127:126]type=2
//  g1: [17:16]data_size=1(2B) | [79:48]tensor_dim0 | [111:80]tensor_dim1
//      | [127:112]tile_dim0 | [143:128]tile_dim1 | [207:160]dim0_stride
//  workgroup_mask=0 (not in a cluster); groups 2/3 zero (<=2D tensor).
// This toolchain exposes the 6-arg builtin form (g0,g1,g2,g3,g4,cpol).
// ---------------------------------------------------------------------------
__device__ __forceinline__ void tdm_load_2d_bf16(unsigned lds_off,
                                                 const void* gptr,
                                                 unsigned tensor_w,
                                                 unsigned tensor_h,
                                                 unsigned row_stride,
                                                 unsigned tile_w,
                                                 unsigned tile_h) {
  const unsigned long long ga = (unsigned long long)(size_t)gptr;
  u32x4 g0;
  g0[0] = 1u;                                       // count=1, user descriptor
  g0[1] = lds_off;                                  // lds_addr (bytes)
  g0[2] = (unsigned)(ga & 0xFFFFFFFFu);             // global_addr[31:0]
  g0[3] = (unsigned)((ga >> 32) & 0x01FFFFFFu) | (2u << 30);  // [56:32]|type=2
  i32x8 g1;
  g1[0] = (int)(1u << 16);                          // data_size=1 (2 bytes)
  g1[1] = (int)((tensor_w & 0xFFFFu) << 16);        // tensor_dim0 lo16
  g1[2] = (int)(((tensor_w >> 16) & 0xFFFFu) | ((tensor_h & 0xFFFFu) << 16));
  g1[3] = (int)(((tensor_h >> 16) & 0xFFFFu) | ((tile_w & 0xFFFFu) << 16));
  g1[4] = (int)(tile_h & 0xFFFFu);                  // tile_dim1 (tile_dim2=0)
  g1[5] = (int)row_stride;                          // tensor_dim0_stride lo32
  g1[6] = 0;
  g1[7] = 0;
  const i32x4 z4 = {0, 0, 0, 0};
  const i32x8 z8 = {0, 0, 0, 0, 0, 0, 0, 0};
  __builtin_amdgcn_tensor_load_to_lds(g0, g1, z4, z4, z8, 0);
}

// ---------------------------------------------------------------------------
// Tiled GEMM: C[M,N] = A[M,K] (fp32 or bf16) * W[K,N] (fp32 -> bf16).
// 256 threads = 8 waves; block tile 64x128; K step 32.
// ---------------------------------------------------------------------------
template <typename AT, bool OUT_BF16>
__global__ __launch_bounds__(256) void gemm_bf16_wmma(
    const AT* __restrict__ A, const float* __restrict__ W,
    void* __restrict__ out, int N, int K) {
  __shared__ __bf16 As[64 * 32];        // row-major 64x32
  __shared__ __bf16 Bsw[8 * 32 * 16];   // fragment-major 32x128

  const int tid = threadIdx.x;
  const int nb = blockIdx.x * 128;
  const int mb = blockIdx.y * 64;
  const int w  = tid >> 5;
  const int mt = (w >> 1) * 16;  // wave row-tile base inside As
  const int nt = (w & 1) * 4;    // wave column-quad: subtiles nt .. nt+3

  const v8f vzero = {0.f, 0.f, 0.f, 0.f, 0.f, 0.f, 0.f, 0.f};
  v8f acc[4] = {vzero, vzero, vzero, vzero};

  for (int kb = 0; kb < K; kb += 32) {
    {  // stage A tile 64x32 row-major (8 elements / thread)
      const int lin = tid * 8;
      const int row = lin >> 5, k = lin & 31;
      stage8(A + (size_t)(mb + row) * K + kb + k, As + row * 32 + k);
    }
    // stage B tile 32x128 into fragment-major layout
    stage_b_swz(W + (size_t)kb * N + nb, N, Bsw, 128, tid);
    __syncthreads();

    const v16bf af = load_a_frag(As + mt * 32, 32);
#pragma unroll
    for (int t = 0; t < 4; ++t) {
      const v16bf bf = load_b_frag_sw(Bsw, nt + t);
      acc[t] = wmma_bf16f32(af, bf, acc[t]);
    }
    __syncthreads();
  }

  // D layout: M = r + 8*(lane>=16), N = lane&15 (CDNA5 ISA C/D table)
  const int lane = tid & 31, half = lane >> 4, nn = lane & 15;
#pragma unroll
  for (int t = 0; t < 4; ++t) {
#pragma unroll
    for (int r = 0; r < 8; ++r) {
      const size_t row = (size_t)(mb + mt + r + 8 * half);
      const size_t col = (size_t)(nb + (nt + t) * 16 + nn);
      if (OUT_BF16)
        ((__bf16*)out)[row * N + col] = (__bf16)acc[t][r];
      else
        ((float*)out)[row * N + col] = acc[t][r];
    }
  }
}

// ---------------------------------------------------------------------------
// Flash attention: one workgroup = (head, 64 query rows), 4 waves of 32.
// K tiles arrive via the Tensor Data Mover; V tiles via swizzled staging.
// Reference semantics: softmax((q.k + causal_mask) * sqrt(2*1024)).
// ---------------------------------------------------------------------------
__global__ __launch_bounds__(128) void flash_attn_wmma(
    const __bf16* __restrict__ Q, const __bf16* __restrict__ Kg,
    const __bf16* __restrict__ V, __bf16* __restrict__ ctx) {
  __shared__ __bf16 Ks[32 * 64];        // row-major [key][hd] (TDM target)
  __shared__ __bf16 Vsw[4 * 32 * 16];   // fragment-major 32x64
  __shared__ __bf16 Ps[4][16 * 32];     // per-wave P bounce (D -> A layout)

  const int tid = threadIdx.x;
  const int w = tid >> 5;
  const int lane = tid & 31, half = lane >> 4, nn = lane & 15;
  const int h = blockIdx.y;
  const int qb = blockIdx.x * 64;
  const int qrow0 = qb + w * 16;
  const float norm = 45.254833995939045f;  // sqrt(2*HIDDEN_DIM)
  const unsigned ksLds = (unsigned)(size_t)&Ks[0];  // LDS byte offset

  // Q tile 16x64 resident in registers (two K=32 A-fragments)
  const __bf16* qptr = Q + (size_t)qrow0 * DMODEL + h * HDIM;
  const v16bf qf0 = load_a_frag(qptr, DMODEL);
  const v16bf qf1 = load_a_frag(qptr + 32, DMODEL);

  float mi[8], li[8];
#pragma unroll
  for (int r = 0; r < 8; ++r) { mi[r] = -INFINITY; li[r] = 0.f; }
  const v8f vzero = {0.f, 0.f, 0.f, 0.f, 0.f, 0.f, 0.f, 0.f};
  v8f acc[4] = {vzero, vzero, vzero, vzero};

  const int nkb = qb / 32 + 2;  // causal bound for this 64-row block
  for (int jb = 0; jb < nkb; ++jb) {
    const int kb = jb * 32;
    // K tile 32(keys) x 64(hd) via TDM, issued by wave 0, overlapping V stage.
    if (w == 0)
      tdm_load_2d_bf16(ksLds, Kg + (size_t)kb * DMODEL + h * HDIM,
                       /*tensor_w=*/DMODEL, /*tensor_h=*/S_LEN,
                       /*row_stride=*/DMODEL, /*tile_w=*/HDIM, /*tile_h=*/32);
    // V tile 32x64 -> fragment-major LDS (all 128 threads)
    stage_b_swz(V + (size_t)kb * DMODEL + h * HDIM, DMODEL, Vsw, 64, tid);
    if (w == 0) __builtin_amdgcn_s_wait_tensorcnt(0);
    __syncthreads();

    // S = Q*K^T : two 16-key subtiles, chained over hd (K-dim 64 = 2x32)
    v8f s0 = vzero, s1 = vzero;
    s0 = wmma_bf16f32(qf0, load_bT_frag(Ks, 64), s0);
    s0 = wmma_bf16f32(qf1, load_bT_frag(Ks + 32, 64), s0);
    s1 = wmma_bf16f32(qf0, load_bT_frag(Ks + 16 * 64, 64), s1);
    s1 = wmma_bf16f32(qf1, load_bT_frag(Ks + 16 * 64 + 32, 64), s1);

    // Online softmax (row r of this lane half <-> M = r + 8*half)
#pragma unroll
    for (int r = 0; r < 8; ++r) {
      const int row = qrow0 + r + 8 * half;
      float v0 = s0[r] * norm;
      float v1 = s1[r] * norm;
      if (kb + nn > row) v0 = -INFINITY;  // causal mask
      if (kb + 16 + nn > row) v1 = -INFINITY;
      const float rm = rmax16(fmaxf(v0, v1));
      const float mnew = fmaxf(mi[r], rm);
      const float p0 = __expf(v0 - mnew);
      const float p1 = __expf(v1 - mnew);
      const float alpha = __expf(mi[r] - mnew);
      li[r] = li[r] * alpha + rsum16(p0 + p1);
      mi[r] = mnew;
#pragma unroll
      for (int t = 0; t < 4; ++t) acc[t][r] *= alpha;
      Ps[w][(r + 8 * half) * 32 + nn] = (__bf16)p0;
      Ps[w][(r + 8 * half) * 32 + 16 + nn] = (__bf16)p1;
    }

    // O += P(16x32) * V(32x64): 4 hd tiles of 16
    const v16bf pf = load_a_frag(Ps[w], 32);
#pragma unroll
    for (int t = 0; t < 4; ++t) {
      const v16bf vf = load_b_frag_sw(Vsw, t);
      acc[t] = wmma_bf16f32(pf, vf, acc[t]);
    }
    __syncthreads();
  }

  // Normalize and write ctx (bf16, [S, H, HD])
#pragma unroll
  for (int r = 0; r < 8; ++r) {
    const float inv = 1.f / li[r];
    const int row = qrow0 + r + 8 * half;
    __bf16* dst = ctx + (size_t)row * DMODEL + h * HDIM;
#pragma unroll
    for (int t = 0; t < 4; ++t) dst[16 * t + nn] = (__bf16)(acc[t][r] * inv);
  }
}

// ---------------------------------------------------------------------------
// Host-side launcher
// ---------------------------------------------------------------------------
extern "C" void kernel_launch(void* const* d_in, const int* in_sizes, int n_in,
                              void* d_out, int out_size, void* d_ws,
                              size_t ws_size, hipStream_t stream) {
  const float* x  = (const float*)d_in[0];
  // d_in[1] is the causal mask; synthesized analytically, never read.
  const float* Wq = (const float*)d_in[2];
  const float* Wk = (const float*)d_in[3];
  const float* Wv = (const float*)d_in[4];
  const float* Wo = (const float*)d_in[5];
  float* out = (float*)d_out;

  const size_t plane = (size_t)S_LEN * DMODEL;  // 4M elements
  __bf16* Qb = (__bf16*)d_ws;
  __bf16* Kb = Qb + plane;
  __bf16* Vb = Kb + plane;
  __bf16* Cb = Vb + plane;  // 32 MB bf16 scratch total

  dim3 gGemm(DMODEL / 128, S_LEN / 64);  // (8, 64)
  gemm_bf16_wmma<float, true><<<gGemm, 256, 0, stream>>>(x, Wq, Qb, DMODEL, EMB);
  gemm_bf16_wmma<float, true><<<gGemm, 256, 0, stream>>>(x, Wk, Kb, DMODEL, EMB);
  gemm_bf16_wmma<float, true><<<gGemm, 256, 0, stream>>>(x, Wv, Vb, DMODEL, EMB);

  dim3 gAttn(S_LEN / 64, NH);  // (64, 16)
  flash_attn_wmma<<<gAttn, 128, 0, stream>>>(Qb, Kb, Vb, Cb);

  gemm_bf16_wmma<__bf16, false><<<gGemm, 256, 0, stream>>>(Cb, Wo, out, EMB, DMODEL);
}
